// TransformerBlock_63617055588884
// MI455X (gfx1250) — compile-verified, matrix operand order checked by reading, not run on hardware
//
#include <hip/hip_runtime.h>

// ---------------------------------------------------------------------------
// Transformer block for gfx1250 (MI455X): f16 WMMA everywhere, flash-attention.
// B=2, S=2048, D=1024, H=16, DH=64, DFF=4096
// ---------------------------------------------------------------------------

#define TB_B   2
#define TB_S   2048
#define TB_D   1024
#define TB_H   16
#define TB_DH  64
#define TB_DFF 4096
#define TB_M   (TB_B * TB_S)   // 4096 rows
#define LN_EPS 1e-5f
#define NEGINF (-1e9f)

typedef __attribute__((ext_vector_type(16))) _Float16 v16h;
typedef __attribute__((ext_vector_type(8)))  _Float16 h8;
typedef __attribute__((ext_vector_type(4)))  _Float16 h4;
typedef __attribute__((ext_vector_type(2)))  __fp16   pk2;   // cvt_pkrtz return type
typedef __attribute__((ext_vector_type(8)))  float    v8f;

union AF16 {
    v16h v;
    h8   h[2];
    _Float16 e[16];
};
union FV4 { float4 v; float f[4]; };

// 8x f32 -> 8x f16 using v_cvt_pk_rtz_f16_f32 (4 instructions)
__device__ inline h8 cvt8(float4 a, float4 b) {
    union { h8 v; pk2 p[4]; } u;
    u.p[0] = __builtin_amdgcn_cvt_pkrtz(a.x, a.y);
    u.p[1] = __builtin_amdgcn_cvt_pkrtz(a.z, a.w);
    u.p[2] = __builtin_amdgcn_cvt_pkrtz(b.x, b.y);
    u.p[3] = __builtin_amdgcn_cvt_pkrtz(b.z, b.w);
    return u.v;
}
__device__ inline h8 cvt8s(float4 a, float4 b, float s) {
    union { h8 v; pk2 p[4]; } u;
    u.p[0] = __builtin_amdgcn_cvt_pkrtz(a.x * s, a.y * s);
    u.p[1] = __builtin_amdgcn_cvt_pkrtz(a.z * s, a.w * s);
    u.p[2] = __builtin_amdgcn_cvt_pkrtz(b.x * s, b.y * s);
    u.p[3] = __builtin_amdgcn_cvt_pkrtz(b.z * s, b.w * s);
    return u.v;
}

// ---------------------------------------------------------------------------
// LayerNorm: one row (D=1024 floats) per 256-thread block.
// ---------------------------------------------------------------------------
__global__ __launch_bounds__(256) void ln_kernel(const float* __restrict__ X,
                                                 const float* __restrict__ gamma,
                                                 const float* __restrict__ beta,
                                                 float* __restrict__ Y) {
    const int row = blockIdx.x;
    const int tid = threadIdx.x;
    const int wid = tid >> 5;
    const int lane = tid & 31;

    const float4* xr = (const float4*)(X + (size_t)row * TB_D);
    float4 v = xr[tid];

    float s = v.x + v.y + v.z + v.w;
    float q = v.x * v.x + v.y * v.y + v.z * v.z + v.w * v.w;
#pragma unroll
    for (int m = 1; m < 32; m <<= 1) {
        s += __shfl_xor(s, m);
        q += __shfl_xor(q, m);
    }

    __shared__ float ssum[8], ssq[8], smu, srstd;
    if (lane == 0) { ssum[wid] = s; ssq[wid] = q; }
    __syncthreads();
    if (tid == 0) {
        float ts = 0.f, tq = 0.f;
#pragma unroll
        for (int i = 0; i < 8; ++i) { ts += ssum[i]; tq += ssq[i]; }
        float mu = ts * (1.0f / TB_D);
        float var = tq * (1.0f / TB_D) - mu * mu;
        smu = mu;
        srstd = rsqrtf(var + LN_EPS);
    }
    __syncthreads();
    const float mu = smu, rstd = srstd;

    float4 g = ((const float4*)gamma)[tid];
    float4 b = ((const float4*)beta)[tid];
    float4 o;
    o.x = (v.x - mu) * rstd * g.x + b.x;
    o.y = (v.y - mu) * rstd * g.y + b.y;
    o.z = (v.z - mu) * rstd * g.z + b.z;
    o.w = (v.w - mu) * rstd * g.w + b.w;
    ((float4*)(Y + (size_t)row * TB_D))[tid] = o;
}

// ---------------------------------------------------------------------------
// Generic WMMA GEMM: C[M,N] = op(A[M,K] @ B[K,N] + bias) (+ resid)
//   HEADED=1 : B element (k,n) at Bsrc[((n>>6)*K + k)*64 + (n&63)] (Wq/Wk/Wv)
// Block = 256 threads (8 waves), 128x128 C tile, K-chunk 64, LDS ping-pong
// double buffering (global loads for tile t+1 issued before WMMA on tile t).
// Each wave: 64x32 sub-tile = 4x2 WMMA accumulators.
// ---------------------------------------------------------------------------
template <int HEADED, int RELU, int BIAS, int RESID>
__global__ __launch_bounds__(256) void gemm_wmma_kernel(
    const float* __restrict__ A, const float* __restrict__ Bsrc,
    const float* __restrict__ bias, const float* __restrict__ resid,
    float* __restrict__ C, int M, int N, int Kd) {

    __shared__ _Float16 As[2][128][64];   // row-major (m, k)   32 KB
    __shared__ _Float16 Bs[2][128][64];   // n-major  (n, k)    32 KB

    const int tid  = threadIdx.x;
    const int wid  = tid >> 5;
    const int lane = tid & 31;
    const int half = lane >> 4;
    const int l16  = lane & 15;

    const int mbase = blockIdx.y * 128;
    const int nbase = blockIdx.x * 128;
    const int waveM = wid >> 2;          // 0..1 -> 64 rows each
    const int waveN = wid & 3;           // 0..3 -> 32 cols each

    v8f acc[4][2];
#pragma unroll
    for (int i = 0; i < 4; ++i)
#pragma unroll
        for (int j = 0; j < 2; ++j)
            acc[i][j] = (v8f){0.f,0.f,0.f,0.f,0.f,0.f,0.f,0.f};

    // staging assignments
    const int arow = tid >> 1;            // 0..127
    const int acb  = (tid & 1) * 32;      // 0 / 32
    const int bk0  = (tid >> 4) * 4;      // 0..60 (k sub-block of 4)
    const int bn0  = (tid & 15) * 8;      // 0..120 (n sub-block of 8)

    float4 ar[8];        // A staging registers (32 floats)
    float4 br[4][2];     // B staging registers (32 floats)

    auto loadA = [&](int kt) {
        const float4* ga =
            (const float4*)(A + (size_t)(mbase + arow) * Kd + kt + acb);
#pragma unroll
        for (int j = 0; j < 8; ++j) ar[j] = ga[j];
    };
    auto loadB = [&](int kt) {
        const int n = nbase + bn0;
#pragma unroll
        for (int kk = 0; kk < 4; ++kk) {
            const int k = kt + bk0 + kk;
            size_t bi = HEADED ? ((size_t)(n >> 6) * Kd + k) * 64 + (n & 63)
                               : (size_t)k * N + n;
            const float4* gb = (const float4*)(Bsrc + bi);
            br[kk][0] = gb[0];
            br[kk][1] = gb[1];
        }
    };
    auto storeA = [&](int p) {
        h8* dst = (h8*)&As[p][arow][acb];
#pragma unroll
        for (int j = 0; j < 4; ++j) dst[j] = cvt8(ar[2 * j], ar[2 * j + 1]);
    };
    auto storeB = [&](int p) {
        float fb[4][8];
#pragma unroll
        for (int kk = 0; kk < 4; ++kk) {
            FV4 a; a.v = br[kk][0];
            FV4 b; b.v = br[kk][1];
#pragma unroll
            for (int c = 0; c < 4; ++c) { fb[kk][c] = a.f[c]; fb[kk][4 + c] = b.f[c]; }
        }
#pragma unroll
        for (int nj = 0; nj < 8; ++nj) {
            union { h4 v; pk2 p[2]; } u;
            u.p[0] = __builtin_amdgcn_cvt_pkrtz(fb[0][nj], fb[1][nj]);
            u.p[1] = __builtin_amdgcn_cvt_pkrtz(fb[2][nj], fb[3][nj]);
            *(h4*)&Bs[p][bn0 + nj][bk0] = u.v;
        }
    };

    // prologue: stage tile 0
    loadA(0); loadB(0);
    storeA(0); storeB(0);
    __syncthreads();

    const int nk = Kd >> 6;
    for (int t = 0; t < nk; ++t) {
        const int p = t & 1;
        const bool hasNext = (t + 1) < nk;
        if (hasNext) { loadA((t + 1) * 64); loadB((t + 1) * 64); }  // in flight

        // ---- WMMA compute on LDS buffer p ----
#pragma unroll
        for (int kc = 0; kc < 64; kc += 32) {
            AF16 af[4], bf[2];
#pragma unroll
            for (int i = 0; i < 4; ++i) {
                const int r = waveM * 64 + i * 16 + l16;
                af[i].h[0] = *(const h8*)&As[p][r][kc + 8 * half];
                af[i].h[1] = *(const h8*)&As[p][r][kc + 16 + 8 * half];
            }
#pragma unroll
            for (int j = 0; j < 2; ++j) {
                const int n = waveN * 32 + j * 16 + l16;
                bf[j].h[0] = *(const h8*)&Bs[p][n][kc + 8 * half];
                bf[j].h[1] = *(const h8*)&Bs[p][n][kc + 16 + 8 * half];
            }
#pragma unroll
            for (int j = 0; j < 2; ++j)
#pragma unroll
                for (int i = 0; i < 4; ++i)
                    acc[i][j] = __builtin_amdgcn_wmma_f32_16x16x32_f16(
                        false, af[i].v, false, bf[j].v, (short)0, acc[i][j],
                        false, false);
        }

        if (hasNext) { storeA(p ^ 1); storeB(p ^ 1); }
        __syncthreads();
    }

    // ---- epilogue: bias / relu / residual (compile-time specialized) ----
#pragma unroll
    for (int i = 0; i < 4; ++i) {
#pragma unroll
        for (int j = 0; j < 2; ++j) {
            const int n = nbase + waveN * 32 + j * 16 + l16;
            const float bv = BIAS ? bias[n] : 0.0f;
#pragma unroll
            for (int e = 0; e < 8; ++e) {
                const int m = mbase + waveM * 64 + i * 16 + e + 8 * half;
                float val = acc[i][j][e] + bv;
                if (RELU) val = fmaxf(val, 0.0f);
                if (RESID) val += resid[(size_t)m * N + n];
                C[(size_t)m * N + n] = val;
            }
        }
    }
}

// ---------------------------------------------------------------------------
// Flash attention: block = (b, h, 128 queries), 8 waves x 16 query rows.
// Q,K,V in [B,S,D] layout (head h at column offset h*64). Causal mask.
// Output written to Attn[B,S,D] (head-major concat == torch.cat order).
// ---------------------------------------------------------------------------
__global__ __launch_bounds__(256) void flash_attn_kernel(
    const float* __restrict__ Qg, const float* __restrict__ Kg,
    const float* __restrict__ Vg, float* __restrict__ Og) {

    __shared__ _Float16 Ks[32][64];       // key-major (t, dh)
    __shared__ _Float16 Vt[64][32];       // dh-major  (dh, t) (transposed)
    __shared__ _Float16 Ps[8][16][32];    // per-wave P scratch (q, t)

    const int tid  = threadIdx.x;
    const int w    = tid >> 5;
    const int lane = tid & 31;
    const int half = lane >> 4;
    const int l16  = lane & 15;

    const int qbase = blockIdx.x * 128;
    const int h     = blockIdx.y;
    const int b     = blockIdx.z;

    const size_t rowstride = TB_D;
    const size_t headoff   = (size_t)h * TB_DH;

    // ---- Q fragments (16 rows x 64, scaled by 1/sqrt(DH)=0.125) ----
    AF16 qf0, qf1;
    {
        const int qr = qbase + w * 16 + l16;
        const float* qp = Qg + ((size_t)(b * TB_S + qr)) * rowstride + headoff;
        const float4* q4 = (const float4*)(qp + 8 * half);
        qf0.h[0] = cvt8s(q4[0], q4[1], 0.125f);       // k:  8h .. 8h+7
        qf0.h[1] = cvt8s(q4[4], q4[5], 0.125f);       // k: 16+8h ..
        qf1.h[0] = cvt8s(q4[8], q4[9], 0.125f);       // k: 32+8h ..
        qf1.h[1] = cvt8s(q4[12], q4[13], 0.125f);     // k: 48+8h ..
    }

    v8f o[4];
#pragma unroll
    for (int j = 0; j < 4; ++j) o[j] = (v8f){0.f,0.f,0.f,0.f,0.f,0.f,0.f,0.f};
    float rm[8], rl[8];
#pragma unroll
    for (int e = 0; e < 8; ++e) { rm[e] = NEGINF; rl[e] = 0.0f; }

    const int qlast = qbase + w * 16 + 15;
    const int nkb = (qbase + 128) >> 5;   // causal: keys < qbase+128

    // staging assignments
    const int kr  = tid >> 3;             // 0..31 key row
    const int kc0 = (tid & 7) * 8;        // 0..56 dh col
    const int vt0 = (tid >> 4) * 2;       // 0..30 key pair
    const int vd0 = (tid & 15) * 4;       // 0..60 dh block of 4

    const v8f zero8 = (v8f){0.f,0.f,0.f,0.f,0.f,0.f,0.f,0.f};

    for (int kb = 0; kb < nkb; ++kb) {
        const int kb32 = kb * 32;
        // ---- cooperative stage: K row-major, V transposed ----
        {
            const float4* gk = (const float4*)(
                Kg + ((size_t)(b * TB_S + kb32 + kr)) * rowstride + headoff + kc0);
            *(h8*)&Ks[kr][kc0] = cvt8(gk[0], gk[1]);

            FV4 va, vb;
            va.v = *(const float4*)(
                Vg + ((size_t)(b * TB_S + kb32 + vt0)) * rowstride + headoff + vd0);
            vb.v = *(const float4*)(
                Vg + ((size_t)(b * TB_S + kb32 + vt0 + 1)) * rowstride + headoff + vd0);
#pragma unroll
            for (int j = 0; j < 4; ++j)
                *(pk2*)&Vt[vd0 + j][vt0] =
                    __builtin_amdgcn_cvt_pkrtz(va.f[j], vb.f[j]);
        }
        __syncthreads();

        if (kb32 <= qlast) {
            // ---- scores: two 16x16 tiles over K-dim 64 (2 chunks of 32) ----
            v8f s0 = zero8, s1 = zero8;
            AF16 kf;
#pragma unroll
            for (int jt = 0; jt < 2; ++jt) {
                const int key = jt * 16 + l16;
                v8f sacc = zero8;
                kf.h[0] = *(const h8*)&Ks[key][8 * half];
                kf.h[1] = *(const h8*)&Ks[key][16 + 8 * half];
                sacc = __builtin_amdgcn_wmma_f32_16x16x32_f16(
                    false, qf0.v, false, kf.v, (short)0, sacc, false, false);
                kf.h[0] = *(const h8*)&Ks[key][32 + 8 * half];
                kf.h[1] = *(const h8*)&Ks[key][48 + 8 * half];
                sacc = __builtin_amdgcn_wmma_f32_16x16x32_f16(
                    false, qf1.v, false, kf.v, (short)0, sacc, false, false);
                if (jt == 0) s0 = sacc; else s1 = sacc;
            }

            // ---- causal mask + online softmax (rows: e + 8*half) ----
            const int t0 = kb32 + l16;
            const int t1 = t0 + 16;
#pragma unroll
            for (int e = 0; e < 8; ++e) {
                const int qr = qbase + w * 16 + e + 8 * half;
                if (t0 > qr) s0[e] = NEGINF;
                if (t1 > qr) s1[e] = NEGINF;

                float mx = fmaxf(s0[e], s1[e]);
#pragma unroll
                for (int m = 1; m < 16; m <<= 1) mx = fmaxf(mx, __shfl_xor(mx, m));
                const float mnew = fmaxf(rm[e], mx);
                const float alpha = __expf(rm[e] - mnew);
                const float p0 = __expf(s0[e] - mnew);
                const float p1 = __expf(s1[e] - mnew);
                float rs = p0 + p1;
#pragma unroll
                for (int m = 1; m < 16; m <<= 1) rs += __shfl_xor(rs, m);
                rl[e] = rl[e] * alpha + rs;
                rm[e] = mnew;
#pragma unroll
                for (int j = 0; j < 4; ++j) o[j][e] *= alpha;

                Ps[w][e + 8 * half][l16]      = (_Float16)p0;
                Ps[w][e + 8 * half][16 + l16] = (_Float16)p1;
            }
            asm volatile("s_wait_dscnt 0" ::: "memory");

            // ---- O += P (16x32) @ V (32x64) ----
            AF16 pf, vf;
            pf.h[0] = *(const h8*)&Ps[w][l16][8 * half];
            pf.h[1] = *(const h8*)&Ps[w][l16][16 + 8 * half];
#pragma unroll
            for (int jt = 0; jt < 4; ++jt) {
                const int dh = jt * 16 + l16;
                vf.h[0] = *(const h8*)&Vt[dh][8 * half];
                vf.h[1] = *(const h8*)&Vt[dh][16 + 8 * half];
                o[jt] = __builtin_amdgcn_wmma_f32_16x16x32_f16(
                    false, pf.v, false, vf.v, (short)0, o[jt], false, false);
            }
        }
        __syncthreads();
    }

    // ---- normalize and write Attn[B,S,D] ----
#pragma unroll
    for (int e = 0; e < 8; ++e) {
        const int qr = qbase + w * 16 + e + 8 * half;
        const float inv = 1.0f / rl[e];
        float* op = Og + ((size_t)(b * TB_S + qr)) * rowstride + headoff;
#pragma unroll
        for (int jt = 0; jt < 4; ++jt) op[jt * 16 + l16] = o[jt][e] * inv;
    }
}

// ---------------------------------------------------------------------------
// Host-side launch
// ---------------------------------------------------------------------------
extern "C" void kernel_launch(void* const* d_in, const int* in_sizes, int n_in,
                              void* d_out, int out_size, void* d_ws, size_t ws_size,
                              hipStream_t stream) {
    (void)in_sizes; (void)n_in; (void)out_size; (void)ws_size;

    const float* X      = (const float*)d_in[0];
    // d_in[1] = attention_mask (causal, hardcoded)
    const float* gamma1 = (const float*)d_in[2];
    const float* beta1  = (const float*)d_in[3];
    const float* Wq     = (const float*)d_in[4];
    const float* bq     = (const float*)d_in[5];
    const float* Wk     = (const float*)d_in[6];
    const float* bk     = (const float*)d_in[7];
    const float* Wv     = (const float*)d_in[8];
    const float* bv     = (const float*)d_in[9];
    const float* W0     = (const float*)d_in[10];
    const float* b0     = (const float*)d_in[11];
    const float* gamma2 = (const float*)d_in[12];
    const float* beta2  = (const float*)d_in[13];
    const float* W1     = (const float*)d_in[14];
    const float* b1     = (const float*)d_in[15];
    const float* W2     = (const float*)d_in[16];
    const float* b2     = (const float*)d_in[17];

    float* ws = (float*)d_ws;
    const size_t SZ = (size_t)TB_M * TB_D;     // 4096*1024
    float* Xn   = ws;                           // reused as Xn2 later
    float* Qb   = ws + 1 * SZ;
    float* Kb   = ws + 2 * SZ;
    float* Vb   = ws + 3 * SZ;
    float* Attn = ws + 4 * SZ;
    float* Y1   = ws + 5 * SZ;
    float* Hf   = ws + 6 * SZ;                  // 4096*4096 = 4*SZ
    float* Out  = (float*)d_out;

    // 1) LN1
    ln_kernel<<<TB_M, 256, 0, stream>>>(X, gamma1, beta1, Xn);

    // 2) QKV projections (headed weight layout [H,D,DH])
    gemm_wmma_kernel<1, 0, 1, 0><<<dim3(TB_D / 128, TB_M / 128), 256, 0, stream>>>(
        Xn, Wq, bq, nullptr, Qb, TB_M, TB_D, TB_D);
    gemm_wmma_kernel<1, 0, 1, 0><<<dim3(TB_D / 128, TB_M / 128), 256, 0, stream>>>(
        Xn, Wk, bk, nullptr, Kb, TB_M, TB_D, TB_D);
    gemm_wmma_kernel<1, 0, 1, 0><<<dim3(TB_D / 128, TB_M / 128), 256, 0, stream>>>(
        Xn, Wv, bv, nullptr, Vb, TB_M, TB_D, TB_D);

    // 3) Flash attention -> Attn[B,S,D]
    flash_attn_kernel<<<dim3(TB_S / 128, TB_H, TB_B), 256, 0, stream>>>(
        Qb, Kb, Vb, Attn);

    // 4) Output projection + residual: Y1 = Attn@W0 + b0 + X
    gemm_wmma_kernel<0, 0, 1, 1><<<dim3(TB_D / 128, TB_M / 128), 256, 0, stream>>>(
        Attn, W0, b0, X, Y1, TB_M, TB_D, TB_D);

    // 5) LN2 (reuse Xn buffer)
    ln_kernel<<<TB_M, 256, 0, stream>>>(Y1, gamma2, beta2, Xn);

    // 6) FFN1: Hf = relu(Xn@W1 + b1)
    gemm_wmma_kernel<0, 1, 1, 0><<<dim3(TB_DFF / 128, TB_M / 128), 256, 0, stream>>>(
        Xn, W1, b1, nullptr, Hf, TB_M, TB_DFF, TB_D);

    // 7) FFN2 + residual: Out = Hf@W2 + b2 + Y1
    gemm_wmma_kernel<0, 0, 1, 1><<<dim3(TB_D / 128, TB_M / 128), 256, 0, stream>>>(
        Hf, W2, b2, Y1, Out, TB_M, TB_D, TB_DFF);
}
